// CausalSelfAttention_43310450213490
// MI455X (gfx1250) — matrix operand order repaired and checked by reading, not compiled, not run on hardware
//
#include <hip/hip_runtime.h>

typedef __attribute__((ext_vector_type(16))) _Float16 v16h;
typedef __attribute__((ext_vector_type(8)))  float    v8f;

#define B_  2
#define T_  2048
#define D_  1024
#define H_  16
#define HD_ 64
#define D3_ 3072

union F16Frag { v16h v; uint4 q[2]; };

__device__ __forceinline__ v8f wmma_f16(const F16Frag& a, const F16Frag& b, v8f c) {
    // D = A(16x32 f16) * B(32x16 f16) + C(16x16 f32)
    return __builtin_amdgcn_wmma_f32_16x16x32_f16(
        /*neg_a=*/false, a.v, /*neg_b=*/false, b.v,
        /*c_mod=*/(short)0, c, /*reuse_a=*/false, /*reuse_b=*/false);
}

// ---------------------------------------------------------------- fp32 -> fp16
__global__ void cvt_f32_to_f16(const float* __restrict__ src,
                               _Float16* __restrict__ dst, int n) {
    int i = blockIdx.x * blockDim.x + threadIdx.x;
    int stride = gridDim.x * blockDim.x;
    for (; i < n; i += stride) dst[i] = (_Float16)src[i];
}

// ------------------------------------------- V slice of qkv -> Vt [B,H,64,T]
__global__ void transpose_v(const _Float16* __restrict__ qkv,
                            _Float16* __restrict__ vt) {
    int idx = blockIdx.x * blockDim.x + threadIdx.x;
    const int total = B_ * H_ * HD_ * T_;
    int stride = gridDim.x * blockDim.x;
    for (; idx < total; idx += stride) {
        int t = idx % T_;
        int d = (idx / T_) % HD_;
        int h = (idx / (T_ * HD_)) % H_;
        int b = idx / (T_ * HD_ * H_);
        vt[idx] = qkv[((size_t)(b * T_ + t)) * D3_ + 2 * D_ + h * HD_ + d];
    }
}

// ------------------------------------------------------------------- GEMM
// C[M,N] = A[M,K] (row major) * W[N,K]^T (row major), f32 accumulate.
// Each wave computes a 32(M) x 64(N) tile (2 A strips x 4 B tiles = 8 WMMAs
// per K-step; each B fragment reused across both A strips -> 1.5 b128 loads
// per WMMA). Block = 256 threads = 8 waves stacked along M (256 rows/block);
// grid = (N/64, M/256).
template <bool OUT_F32>
__global__ __launch_bounds__(256) void gemm_wmma(const _Float16* __restrict__ A,
                                                 const _Float16* __restrict__ W,
                                                 void* __restrict__ out,
                                                 int M, int N, int K) {
    const int lane = threadIdx.x & 31;
    const int wave = threadIdx.x >> 5;
    const int m16  = lane & 15;
    const int hw   = lane >> 4;
    const int hw8  = hw * 8;    // C-layout row offset (M = r + 8*hw)
    const int hw16 = hw * 16;   // B-operand K base
    const int row0 = (blockIdx.y * 8 + wave) * 32;
    const int col0 = blockIdx.x * 64;

    const _Float16* aRow0 = A + (size_t)(row0 +  0 + m16) * K;
    const _Float16* aRow1 = A + (size_t)(row0 + 16 + m16) * K;
    const _Float16* wRow0 = W + (size_t)(col0 +  0 + m16) * K;
    const _Float16* wRow1 = W + (size_t)(col0 + 16 + m16) * K;
    const _Float16* wRow2 = W + (size_t)(col0 + 32 + m16) * K;
    const _Float16* wRow3 = W + (size_t)(col0 + 48 + m16) * K;

    v8f acc00 = {}, acc01 = {}, acc02 = {}, acc03 = {};   // rows row0..row0+15
    v8f acc10 = {}, acc11 = {}, acc12 = {}, acc13 = {};   // rows row0+16..+31

    for (int k = 0; k < K; k += 32) {
        if (k + 256 < K) {
            __builtin_prefetch(aRow0 + k + 256, 0, 1);
            __builtin_prefetch(aRow1 + k + 256, 0, 1);
            __builtin_prefetch(wRow0 + k + 256, 0, 1);
            __builtin_prefetch(wRow2 + k + 256, 0, 1);
        }
        F16Frag a0, a1;
        a0.q[0] = *reinterpret_cast<const uint4*>(aRow0 + k + hw16);
        a0.q[1] = *reinterpret_cast<const uint4*>(aRow0 + k + hw16 + 8);
        a1.q[0] = *reinterpret_cast<const uint4*>(aRow1 + k + hw16);
        a1.q[1] = *reinterpret_cast<const uint4*>(aRow1 + k + hw16 + 8);

        F16Frag b;
        b.q[0] = *reinterpret_cast<const uint4*>(wRow0 + k + hw16);
        b.q[1] = *reinterpret_cast<const uint4*>(wRow0 + k + hw16 + 8);
        acc00 = wmma_f16(a0, b, acc00);
        acc10 = wmma_f16(a1, b, acc10);
        b.q[0] = *reinterpret_cast<const uint4*>(wRow1 + k + hw16);
        b.q[1] = *reinterpret_cast<const uint4*>(wRow1 + k + hw16 + 8);
        acc01 = wmma_f16(a0, b, acc01);
        acc11 = wmma_f16(a1, b, acc11);
        b.q[0] = *reinterpret_cast<const uint4*>(wRow2 + k + hw16);
        b.q[1] = *reinterpret_cast<const uint4*>(wRow2 + k + hw16 + 8);
        acc02 = wmma_f16(a0, b, acc02);
        acc12 = wmma_f16(a1, b, acc12);
        b.q[0] = *reinterpret_cast<const uint4*>(wRow3 + k + hw16);
        b.q[1] = *reinterpret_cast<const uint4*>(wRow3 + k + hw16 + 8);
        acc03 = wmma_f16(a0, b, acc03);
        acc13 = wmma_f16(a1, b, acc13);
    }

#pragma unroll
    for (int r = 0; r < 8; ++r) {
        const int rowA = row0 + r + hw8;            // C layout: vgpr r -> M=r / r+8
        const int rowB = row0 + 16 + r + hw8;
        const size_t baseA = (size_t)rowA * N + col0 + m16;
        const size_t baseB = (size_t)rowB * N + col0 + m16;
        if (OUT_F32) {
            float* o = (float*)out;
            o[baseA]      = acc00[r];
            o[baseA + 16] = acc01[r];
            o[baseA + 32] = acc02[r];
            o[baseA + 48] = acc03[r];
            o[baseB]      = acc10[r];
            o[baseB + 16] = acc11[r];
            o[baseB + 32] = acc12[r];
            o[baseB + 48] = acc13[r];
        } else {
            _Float16* o = (_Float16*)out;
            o[baseA]      = (_Float16)acc00[r];
            o[baseA + 16] = (_Float16)acc01[r];
            o[baseA + 32] = (_Float16)acc02[r];
            o[baseA + 48] = (_Float16)acc03[r];
            o[baseB]      = (_Float16)acc10[r];
            o[baseB + 16] = (_Float16)acc11[r];
            o[baseB + 32] = (_Float16)acc12[r];
            o[baseB + 48] = (_Float16)acc13[r];
        }
    }
}

// --------------------------------------------------------- flash attention
// One wave (32 threads) per (b, h, 16-row q tile). Online softmax over key
// chunks of 32. P goes through LDS to convert C-layout -> A-operand layout.
__global__ __launch_bounds__(32) void attn_flash(const _Float16* __restrict__ qkv,
                                                 const _Float16* __restrict__ vt,
                                                 _Float16* __restrict__ y) {
    __shared__ __align__(16) _Float16 Pl[16 * 32];

    const int lane = threadIdx.x & 31;
    const int m16  = lane & 15;
    const int hw   = lane >> 4;
    const int hw8  = hw * 8;
    const int hw16 = hw * 16;
    const int qt = blockIdx.x;
    const int h  = blockIdx.y;
    const int b  = blockIdx.z;
    const int q0 = qt * 16;

    const _Float16* qkvB  = qkv + (size_t)b * T_ * D3_;
    const _Float16* qRow  = qkvB + (size_t)(q0 + m16) * D3_ + h * HD_;
    const _Float16* kBase = qkvB + D_ + h * HD_;
    const _Float16* vtB   = vt + ((size_t)(b * H_ + h)) * HD_ * T_;

    // Q as two 16x32 A-operands (d = 0..31, 32..63)
    F16Frag qa0, qa1;
    qa0.q[0] = *reinterpret_cast<const uint4*>(qRow + hw16);
    qa0.q[1] = *reinterpret_cast<const uint4*>(qRow + hw16 + 8);
    qa1.q[0] = *reinterpret_cast<const uint4*>(qRow + 32 + hw16);
    qa1.q[1] = *reinterpret_cast<const uint4*>(qRow + 32 + hw16 + 8);

    v8f o0 = {}, o1 = {}, o2 = {}, o3 = {};
    float mrow[8], lrow[8];
#pragma unroll
    for (int r = 0; r < 8; ++r) { mrow[r] = -1e30f; lrow[r] = 0.0f; }

    const float scale = 0.125f;  // 1/sqrt(64)

    for (int kb = 0; kb < q0 + 16; kb += 32) {
        // ---- S = Q K^T for 32 keys (two 16x16 tiles) ----
        v8f s0 = {}, s1 = {};
        const _Float16* k0p = kBase + (size_t)(kb + m16) * D3_;
        const _Float16* k1p = kBase + (size_t)(kb + 16 + m16) * D3_;
        F16Frag kf;
        kf.q[0] = *reinterpret_cast<const uint4*>(k0p + hw16);
        kf.q[1] = *reinterpret_cast<const uint4*>(k0p + hw16 + 8);
        s0 = wmma_f16(qa0, kf, s0);
        kf.q[0] = *reinterpret_cast<const uint4*>(k0p + 32 + hw16);
        kf.q[1] = *reinterpret_cast<const uint4*>(k0p + 32 + hw16 + 8);
        s0 = wmma_f16(qa1, kf, s0);
        kf.q[0] = *reinterpret_cast<const uint4*>(k1p + hw16);
        kf.q[1] = *reinterpret_cast<const uint4*>(k1p + hw16 + 8);
        s1 = wmma_f16(qa0, kf, s1);
        kf.q[0] = *reinterpret_cast<const uint4*>(k1p + 32 + hw16);
        kf.q[1] = *reinterpret_cast<const uint4*>(k1p + 32 + hw16 + 8);
        s1 = wmma_f16(qa1, kf, s1);

        // ---- causal mask + online softmax (row = r + hw8, col = lane&15) ----
        const int kcol0 = kb + m16;
        const int kcol1 = kb + 16 + m16;
#pragma unroll
        for (int r = 0; r < 8; ++r) {
            const int qidx = q0 + r + hw8;
            float v0 = (kcol0 <= qidx) ? s0[r] * scale : -1e30f;
            float v1 = (kcol1 <= qidx) ? s1[r] * scale : -1e30f;
            float mx = fmaxf(v0, v1);
            mx = fmaxf(mx, __shfl_xor(mx, 1));
            mx = fmaxf(mx, __shfl_xor(mx, 2));
            mx = fmaxf(mx, __shfl_xor(mx, 4));
            mx = fmaxf(mx, __shfl_xor(mx, 8));
            const float mnew = fmaxf(mrow[r], mx);
            const float corr = __expf(mrow[r] - mnew);
            const float p0 = __expf(v0 - mnew);
            const float p1 = __expf(v1 - mnew);
            float rs = p0 + p1;
            rs += __shfl_xor(rs, 1);
            rs += __shfl_xor(rs, 2);
            rs += __shfl_xor(rs, 4);
            rs += __shfl_xor(rs, 8);
            lrow[r] = lrow[r] * corr + rs;
            mrow[r] = mnew;
            o0[r] *= corr; o1[r] *= corr; o2[r] *= corr; o3[r] *= corr;
            Pl[(r + hw8) * 32 + m16]      = (_Float16)p0;
            Pl[(r + hw8) * 32 + 16 + m16] = (_Float16)p1;
        }
        __syncthreads();  // single-wave WG: orders P store -> load (ds wait)

        // ---- O += P V  (P: 16x32 A-operand from LDS, V^T rows contiguous) ----
        F16Frag pa;
        pa.q[0] = *reinterpret_cast<const uint4*>(&Pl[m16 * 32 + hw16]);
        pa.q[1] = *reinterpret_cast<const uint4*>(&Pl[m16 * 32 + hw16 + 8]);

        F16Frag vb;
        const _Float16* vr = vtB + (size_t)(0 + m16) * T_ + kb + hw16;
        vb.q[0] = *reinterpret_cast<const uint4*>(vr);
        vb.q[1] = *reinterpret_cast<const uint4*>(vr + 8);
        o0 = wmma_f16(pa, vb, o0);
        vr = vtB + (size_t)(16 + m16) * T_ + kb + hw16;
        vb.q[0] = *reinterpret_cast<const uint4*>(vr);
        vb.q[1] = *reinterpret_cast<const uint4*>(vr + 8);
        o1 = wmma_f16(pa, vb, o1);
        vr = vtB + (size_t)(32 + m16) * T_ + kb + hw16;
        vb.q[0] = *reinterpret_cast<const uint4*>(vr);
        vb.q[1] = *reinterpret_cast<const uint4*>(vr + 8);
        o2 = wmma_f16(pa, vb, o2);
        vr = vtB + (size_t)(48 + m16) * T_ + kb + hw16;
        vb.q[0] = *reinterpret_cast<const uint4*>(vr);
        vb.q[1] = *reinterpret_cast<const uint4*>(vr + 8);
        o3 = wmma_f16(pa, vb, o3);
        __syncthreads();  // protect Pl before next chunk's stores
    }

    // ---- normalize and write y (f16) ----
    const size_t yBase = (size_t)b * T_ * D_;
#pragma unroll
    for (int r = 0; r < 8; ++r) {
        const float inv = 1.0f / lrow[r];
        const int row = q0 + r + hw8;
        const size_t base = yBase + (size_t)row * D_ + h * HD_ + m16;
        y[base]      = (_Float16)(o0[r] * inv);
        y[base + 16] = (_Float16)(o1[r] * inv);
        y[base + 32] = (_Float16)(o2[r] * inv);
        y[base + 48] = (_Float16)(o3[r] * inv);
    }
}

// ---------------------------------------------------------------- launcher
extern "C" void kernel_launch(void* const* d_in, const int* in_sizes, int n_in,
                              void* d_out, int out_size, void* d_ws, size_t ws_size,
                              hipStream_t stream) {
    (void)in_sizes; (void)n_in; (void)out_size; (void)ws_size;

    const float* x  = (const float*)d_in[0];   // [B,T,D]
    const float* wq = (const float*)d_in[1];   // [3D,D]
    const float* wp = (const float*)d_in[2];   // [D,D]

    const size_t NXB = (size_t)B_ * T_ * D_;       // 4,194,304
    const size_t NWQ = (size_t)3 * D_ * D_;        // 3,145,728
    const size_t NWP = (size_t)D_ * D_;            // 1,048,576
    const size_t NQK = (size_t)B_ * T_ * D3_;      // 12,582,912
    const size_t NVT = (size_t)B_ * H_ * HD_ * T_; // 4,194,304

    _Float16* xb  = (_Float16*)d_ws;
    _Float16* wqb = xb  + NXB;
    _Float16* wpb = wqb + NWQ;
    _Float16* qkb = wpb + NWP;
    _Float16* vtb = qkb + NQK;
    _Float16* yb  = vtb + NVT;   // total ~56 MB of d_ws

    cvt_f32_to_f16<<<dim3(2048), dim3(256), 0, stream>>>(x,  xb,  (int)NXB);
    cvt_f32_to_f16<<<dim3(2048), dim3(256), 0, stream>>>(wq, wqb, (int)NWQ);
    cvt_f32_to_f16<<<dim3(1024), dim3(256), 0, stream>>>(wp, wpb, (int)NWP);

    // qkv = x * w_qkv^T  : M=4096, N=3072, K=1024, f16 out
    gemm_wmma<false><<<dim3(D3_ / 64, (B_ * T_) / 256), dim3(256), 0, stream>>>(
        xb, wqb, (void*)qkb, B_ * T_, D3_, D_);

    transpose_v<<<dim3(4096), dim3(256), 0, stream>>>(qkb, vtb);

    attn_flash<<<dim3(T_ / 16, H_, B_), dim3(32), 0, stream>>>(qkb, vtb, yb);

    // out = y * w_proj^T : M=4096, N=1024, K=1024, f32 out
    gemm_wmma<true><<<dim3(D_ / 64, (B_ * T_) / 256), dim3(256), 0, stream>>>(
        yb, wpb, d_out, B_ * T_, D_, D_);
}
